// calculate_forces_6571299963157
// MI455X (gfx1250) — compile-verified, matrix operand order checked by reading, not run on hardware
//
#include <hip/hip_runtime.h>

// ---------------------------------------------------------------------------
// Problem constants (match reference)
// ---------------------------------------------------------------------------
#define BB   4
#define NN   8192
#define KNNK 20
#define DIMF 32
#define XSTR 64          // padded feature row stride (elements)

typedef __attribute__((ext_vector_type(16))) _Float16 v16h;
typedef __attribute__((ext_vector_type(8)))  _Float16 v8h;
typedef __attribute__((ext_vector_type(8)))  float    v8f;

static __device__ __forceinline__ v8f wmma_f16(v16h a, v16h b, v8f c) {
  // D = A(16x32 f16) x B(32x16 f16) + C(16x16 f32)
  return __builtin_amdgcn_wmma_f32_16x16x32_f16(false, a, false, b, (short)0, c,
                                                false, false);
}

// A-fragment element j (0..15) of lane (h = lane>>4) maps to
//   K = h*8 + j          (j < 8)
//   K = 8 + h*8 + j      (j >= 8)
// i.e. two contiguous 8-half (16B) segments per lane: [h*8, +8) and [16+h*8, +8)

// Build A fragment = (x[col] - x[row]) for one 32-wide K chunk.
static __device__ __forceinline__ v16h load_a_diff(const _Float16* __restrict__ xc,
                                                   const _Float16* __restrict__ xr,
                                                   int koff, int h) {
  v16h a;
#pragma unroll
  for (int s = 0; s < 2; ++s) {
    const int k0 = koff + s * 16 + h * 8;
    const v8h c8 = *(const v8h*)(xc + k0);
    const v8h r8 = *(const v8h*)(xr + k0);
#pragma unroll
    for (int j = 0; j < 8; ++j)
      a[s * 8 + j] = (_Float16)((float)c8[j] - (float)r8[j]);
  }
  return a;
}

// Load A fragment from a row-major 16 x XSTR f16 LDS tile.
static __device__ __forceinline__ v16h load_a_lds(const _Float16* __restrict__ lds,
                                                  int koff, int lane) {
  const int m = lane & 15, h = lane >> 4;
  v16h a;
#pragma unroll
  for (int s = 0; s < 2; ++s) {
    const v8h d = *(const v8h*)(lds + m * XSTR + koff + s * 16 + h * 8);
#pragma unroll
    for (int j = 0; j < 8; ++j) a[s * 8 + j] = d[j];
  }
  return a;
}

// Load A fragment from a row-major f32 buffer (agg), converting to f16.
static __device__ __forceinline__ v16h load_a_f32(const float* __restrict__ rowp,
                                                  int koff, int h) {
  v16h a;
#pragma unroll
  for (int s = 0; s < 2; ++s) {
    const int k0 = koff + s * 16 + h * 8;
    const float4 f0 = *(const float4*)(rowp + k0);
    const float4 f1 = *(const float4*)(rowp + k0 + 4);
    a[s * 8 + 0] = (_Float16)f0.x; a[s * 8 + 1] = (_Float16)f0.y;
    a[s * 8 + 2] = (_Float16)f0.z; a[s * 8 + 3] = (_Float16)f0.w;
    a[s * 8 + 4] = (_Float16)f1.x; a[s * 8 + 5] = (_Float16)f1.y;
    a[s * 8 + 6] = (_Float16)f1.z; a[s * 8 + 7] = (_Float16)f1.w;
  }
  return a;
}

// C layout: lane holds column n = lane&15; VGPR r holds row m = r + 8*(lane>>4).
static __device__ __forceinline__ void store_c_relu_lds(_Float16* __restrict__ lds,
                                                        v8f c, int ntile, int lane) {
  const int n = lane & 15, h = lane >> 4;
#pragma unroll
  for (int r = 0; r < 8; ++r) {
    float v = c[r];
    v = v > 0.f ? v : 0.f;
    lds[(r + 8 * h) * XSTR + ntile * 16 + n] = (_Float16)v;
  }
}

static __device__ __forceinline__ v8f bias_c(const float* __restrict__ b, int ntile,
                                             int lane, int dreal) {
  const int n = ntile * 16 + (lane & 15);
  const float bv = (n < dreal) ? b[n] : 0.f;
  v8f c;
#pragma unroll
  for (int r = 0; r < 8; ++r) c[r] = bv;
  return c;
}

// ---------------------------------------------------------------------------
// Weight prep: swizzle f32 [K,N] weights into per-lane contiguous B fragments
// out[((t*CH + c)*32 + lane)*16 + j]  (one 32B v16h load per WMMA in hot loop)
// ---------------------------------------------------------------------------
static __device__ void swizzle_w(const float* __restrict__ w, int Kd, int Nd,
                                 _Float16* __restrict__ outw, int tid) {
  const int CH = (Kd > 32) ? 2 : 1;
  const int NT = (Nd + 15) / 16;
  const int total = NT * CH * 32;
  for (int idx = tid; idx < total; idx += 256) {
    const int lane = idx & 31;
    const int c = (idx >> 5) % CH;
    const int t = idx / (CH * 32);
    const int hh = lane >> 4;
    const int nn2 = t * 16 + (lane & 15);
#pragma unroll
    for (int j = 0; j < 16; ++j) {
      const int kk2 = c * 32 + ((j < 8) ? (hh * 8 + j) : (8 + hh * 8 + j));
      float v = 0.f;
      if (kk2 < Kd && nn2 < Nd) v = w[kk2 * Nd + nn2];
      outw[idx * 16 + j] = (_Float16)v;
    }
  }
}

__global__ __launch_bounds__(256)
void prep_layer_kernel(const float* __restrict__ w1m1, const float* __restrict__ b1m1,
                       const float* __restrict__ w2m1, const float* __restrict__ b2m1,
                       const float* __restrict__ w1m2, const float* __restrict__ w2m2,
                       int din,
                       _Float16* w1m1s, _Float16* w2m1s,
                       _Float16* w1m2s, _Float16* w2m2s,
                       float* __restrict__ cvec) {
  const int tid = threadIdx.x;
  swizzle_w(w1m1, din, din, w1m1s, tid);
  swizzle_w(w2m1, din, din, w2m1s, tid);
  swizzle_w(w1m2, din, 32, w1m2s, tid);
  swizzle_w(w2m2, 32, 32, w2m2s, tid);
  // self-loop message constant: mlp1(0) = relu(b1) @ W2 + b2 (zero padded)
  if (tid < XSTR) {
    float v = 0.f;
    if (tid < din) {
      v = b2m1[tid];
      for (int k2 = 0; k2 < din; ++k2) {
        const float r = b1m1[k2] > 0.f ? b1m1[k2] : 0.f;
        v += r * w2m1[k2 * din + tid];
      }
    }
    cvec[tid] = v;
  }
}

// ---------------------------------------------------------------------------
// Feature build: xh[b,n,0:32]=emb[atoms[n]], [32:35]=pos, rest zero (f16)
// ---------------------------------------------------------------------------
__global__ void build_x_kernel(const float* __restrict__ pos,
                               const int* __restrict__ atoms,
                               const float* __restrict__ emb,
                               _Float16* __restrict__ xh) {
  const int i = blockIdx.x * blockDim.x + threadIdx.x;
  if (i >= BB * NN * XSTR) return;
  const int k = i & (XSTR - 1);
  const int n = (i >> 6) % NN;
  const int b = i / (NN * XSTR);
  float v = 0.f;
  if (k < DIMF) v = emb[atoms[n] * DIMF + k];
  else if (k < DIMF + 3) v = pos[((size_t)b * NN + n) * 3 + (k - DIMF)];
  xh[i] = (_Float16)v;
}

// ---------------------------------------------------------------------------
// KNN top-20 (smallest squared distance), LDS-blocked candidates,
// register-resident running top-k per query thread.
// ---------------------------------------------------------------------------
__global__ __launch_bounds__(256)
void knn_kernel(const float* __restrict__ pos, int* __restrict__ knn) {
  __shared__ float cp[256 * 3];
  const int b = blockIdx.y;
  const int q = blockIdx.x * 256 + threadIdx.x;
  const float* pb = pos + (size_t)b * NN * 3;
  const float qx = pb[q * 3 + 0], qy = pb[q * 3 + 1], qz = pb[q * 3 + 2];
  float bd[KNNK];
  int bi[KNNK];
#pragma unroll
  for (int j = 0; j < KNNK; ++j) { bd[j] = 3.4e38f; bi[j] = 0; }
  float worst = 3.4e38f;
  for (int c0 = 0; c0 < NN; c0 += 256) {
    __syncthreads();
    const int ci = c0 + threadIdx.x;
    cp[threadIdx.x * 3 + 0] = pb[ci * 3 + 0];
    cp[threadIdx.x * 3 + 1] = pb[ci * 3 + 1];
    cp[threadIdx.x * 3 + 2] = pb[ci * 3 + 2];
    __syncthreads();
    for (int c = 0; c < 256; ++c) {
      const float dx = qx - cp[c * 3 + 0];
      const float dy = qy - cp[c * 3 + 1];
      const float dz = qz - cp[c * 3 + 2];
      const float d2 = dx * dx + dy * dy + dz * dz;
      if (d2 < worst) {
        int slot = 0; float mv = bd[0];
#pragma unroll
        for (int j = 1; j < KNNK; ++j) if (bd[j] > mv) { mv = bd[j]; slot = j; }
#pragma unroll
        for (int j = 0; j < KNNK; ++j) if (j == slot) { bd[j] = d2; bi[j] = c0 + c; }
        float w2 = bd[0];
#pragma unroll
        for (int j = 1; j < KNNK; ++j) w2 = fmaxf(w2, bd[j]);
        worst = w2;
      }
    }
  }
  int* kq = knn + ((size_t)b * NN + q) * KNNK;
#pragma unroll
  for (int j = 0; j < KNNK; ++j) kq[j] = bi[j];
}

// ---------------------------------------------------------------------------
// agg init = self-loop constant (cols < din), zero in the pad region
// ---------------------------------------------------------------------------
__global__ void init_agg_kernel(float* __restrict__ agg,
                                const float* __restrict__ cvec, int din) {
  const int i = blockIdx.x * blockDim.x + threadIdx.x;
  if (i >= BB * NN * XSTR) return;
  const int k = i & (XSTR - 1);
  agg[i] = (k < din) ? cvec[k] : 0.f;
}

// ---------------------------------------------------------------------------
// Edge kernel: one wave per 16-edge tile.
//   diff = x[col]-x[row] -> WMMA (din->din) -> relu -> LDS bounce ->
//   WMMA (din->din) -> atomic scatter-add into agg[row]
// ---------------------------------------------------------------------------
template <int DIN, bool IS_KNN>
__global__ __launch_bounds__(256)
void gin_edge_kernel(const _Float16* __restrict__ xh,
                     const int* __restrict__ knn,
                     const int* __restrict__ bonds,
                     int ntiles_e,
                     const _Float16* __restrict__ w1sw, const float* __restrict__ b1,
                     const _Float16* __restrict__ w2sw, const float* __restrict__ b2,
                     float* __restrict__ agg) {
  constexpr int CH = (DIN > 32) ? 2 : 1;     // K chunks of 32
  constexpr int NT = (DIN + 15) / 16;        // 16-wide N tiles
  __shared__ _Float16 lds[8][16 * XSTR];
  __shared__ int rowi[8][16];
  const int wv = threadIdx.x >> 5;
  const int lane = threadIdx.x & 31;
  const int b = blockIdx.y;
  const int h = lane >> 4;
  const int m = lane & 15;

  // zero pad columns once (GEMM2 K-chunk 2 reads up to col 63)
  for (int i = lane; i < 16 * XSTR; i += 32) lds[wv][i] = (_Float16)0.f;
  asm volatile("s_wait_dscnt 0x0" ::: "memory");

  const _Float16* xb = xh + (size_t)b * NN * XSTR;
  float* aggb = agg + (size_t)b * NN * XSTR;

  for (int t = blockIdx.x * 8 + wv; t < ntiles_e; t += gridDim.x * 8) {
    const int e = t * 16 + m;
    int row, col;
    if (IS_KNN) {
      row = e / KNNK;
      col = knn[(size_t)b * NN * KNNK + e];
    } else {
      row = bonds[2 * e];
      col = bonds[2 * e + 1];
    }
    rowi[wv][m] = row;  // lanes 16-31 duplicate same values (benign)

    const _Float16* xr = xb + (size_t)row * XSTR;
    const _Float16* xc = xb + (size_t)col * XSTR;

    v16h af[CH];
#pragma unroll
    for (int c = 0; c < CH; ++c) af[c] = load_a_diff(xc, xr, c * 32, h);

    // GEMM1: relu(diff @ W1 + b1) -> LDS
#pragma unroll
    for (int nt = 0; nt < NT; ++nt) {
      v8f acc = bias_c(b1, nt, lane, DIN);
#pragma unroll
      for (int c = 0; c < CH; ++c) {
        const v16h bw = *(const v16h*)(w1sw + ((size_t)(nt * CH + c) * 32 + lane) * 16);
        acc = wmma_f16(af[c], bw, acc);
      }
      store_c_relu_lds(&lds[wv][0], acc, nt, lane);
    }
    asm volatile("s_wait_dscnt 0x0" ::: "memory");

    v16h a2[CH];
#pragma unroll
    for (int c = 0; c < CH; ++c) a2[c] = load_a_lds(&lds[wv][0], c * 32, lane);

    // GEMM2: msg = h1 @ W2 + b2; scatter-add to agg[row]
#pragma unroll
    for (int nt = 0; nt < NT; ++nt) {
      v8f acc = bias_c(b2, nt, lane, DIN);
#pragma unroll
      for (int c = 0; c < CH; ++c) {
        const v16h bw = *(const v16h*)(w2sw + ((size_t)(nt * CH + c) * 32 + lane) * 16);
        acc = wmma_f16(a2[c], bw, acc);
      }
      const int colg = nt * 16 + m;
      if (colg < DIN) {
#pragma unroll
        for (int r = 0; r < 8; ++r) {
          const int node = rowi[wv][r + 8 * h];
          atomicAdd(&aggb[(size_t)node * XSTR + colg], acc[r]);
        }
      }
    }
  }
}

// ---------------------------------------------------------------------------
// Node kernel: x_next = relu( relu(agg @ W1 + b1) @ W2 + b2 )  (mlp2 + GIN relu)
// One wave per 16-node tile; coalesced f16 output through LDS.
// ---------------------------------------------------------------------------
template <int DIN>
__global__ __launch_bounds__(256)
void gin_node_kernel(const float* __restrict__ agg,
                     const _Float16* __restrict__ w1sw, const float* __restrict__ b1,
                     const _Float16* __restrict__ w2sw, const float* __restrict__ b2,
                     _Float16* __restrict__ xout) {
  constexpr int CH1 = (DIN > 32) ? 2 : 1;
  constexpr int NT = 2;  // HID = 32
  __shared__ _Float16 lds[8][16 * XSTR];
  const int wv = threadIdx.x >> 5;
  const int lane = threadIdx.x & 31;
  const int b = blockIdx.y;
  const int h = lane >> 4;
  const int m = lane & 15;

  for (int i = lane; i < 16 * XSTR; i += 32) lds[wv][i] = (_Float16)0.f;
  asm volatile("s_wait_dscnt 0x0" ::: "memory");

  const float* aggb = agg + (size_t)b * NN * XSTR;
  _Float16* xb = xout + (size_t)b * NN * XSTR;

  const int ntiles = NN / 16;
  for (int t = blockIdx.x * 8 + wv; t < ntiles; t += gridDim.x * 8) {
    const int node0 = t * 16;
    v16h af[CH1];
#pragma unroll
    for (int c = 0; c < CH1; ++c)
      af[c] = load_a_f32(aggb + (size_t)(node0 + m) * XSTR, c * 32, h);

#pragma unroll
    for (int nt = 0; nt < NT; ++nt) {
      v8f acc = bias_c(b1, nt, lane, 32);
#pragma unroll
      for (int c = 0; c < CH1; ++c) {
        const v16h bw = *(const v16h*)(w1sw + ((size_t)(nt * CH1 + c) * 32 + lane) * 16);
        acc = wmma_f16(af[c], bw, acc);
      }
      store_c_relu_lds(&lds[wv][0], acc, nt, lane);
    }
    asm volatile("s_wait_dscnt 0x0" ::: "memory");

    const v16h a2 = load_a_lds(&lds[wv][0], 0, lane);

#pragma unroll
    for (int nt = 0; nt < NT; ++nt) {
      v8f acc = bias_c(b2, nt, lane, 32);
      const v16h bw = *(const v16h*)(w2sw + ((size_t)nt * 32 + lane) * 16);
      acc = wmma_f16(a2, bw, acc);
      store_c_relu_lds(&lds[wv][0], acc, nt, lane);  // GIN relu folded in
    }
    asm volatile("s_wait_dscnt 0x0" ::: "memory");

    // coalesced write of the 16x32 result tile (lane -> row m, 16-half seg h)
    const v8h* src = (const v8h*)(&lds[wv][m * XSTR + h * 16]);
    v8h* dst = (v8h*)(xb + (size_t)(node0 + m) * XSTR + h * 16);
    dst[0] = src[0];
    dst[1] = src[1];
  }
}

// ---------------------------------------------------------------------------
// Final projection: out(=|+=) x @ fc_w + fc_b   (32 -> 3, trivial VALU)
// ---------------------------------------------------------------------------
template <bool ADD>
__global__ void fc_kernel(const _Float16* __restrict__ xh,
                          const float* __restrict__ fcw,
                          const float* __restrict__ fcb,
                          float* __restrict__ out) {
  const int i = blockIdx.x * blockDim.x + threadIdx.x;
  if (i >= BB * NN) return;
  const _Float16* xr = xh + (size_t)i * XSTR;
  float o0 = fcb[0], o1 = fcb[1], o2 = fcb[2];
#pragma unroll
  for (int k = 0; k < 32; ++k) {
    const float x = (float)xr[k];
    o0 += x * fcw[k * 3 + 0];
    o1 += x * fcw[k * 3 + 1];
    o2 += x * fcw[k * 3 + 2];
  }
  float* op = out + (size_t)i * 3;
  if (ADD) { op[0] += o0; op[1] += o1; op[2] += o2; }
  else     { op[0] = o0;  op[1] = o1;  op[2] = o2; }
}

// ---------------------------------------------------------------------------
// Workspace layout (bytes) ~23.8 MB total
// ---------------------------------------------------------------------------
static const size_t OFF_KNN = 0;                                   // 2,621,440
static const size_t OFF_XH0 = 2621440;                             // 4,194,304
static const size_t OFF_XA  = OFF_XH0 + 4194304;                   // 4,194,304
static const size_t OFF_XB  = OFF_XA + 4194304;                    // 4,194,304
static const size_t OFF_AGG = OFF_XB + 4194304;                    // 8,388,608
static const size_t OFF_W   = OFF_AGG + 8388608;                   // 6 * 32768

extern "C" void kernel_launch(void* const* d_in, const int* in_sizes, int n_in,
                              void* d_out, int out_size, void* d_ws, size_t ws_size,
                              hipStream_t stream) {
  (void)in_sizes; (void)n_in; (void)out_size; (void)ws_size;
  const float* pos   = (const float*)d_in[0];
  const int*   atoms = (const int*)d_in[1];
  const int*   bonds = (const int*)d_in[2];
  const float* emb   = (const float*)d_in[3];

  // gin param leaves (pytree order: alphabetical keys within each dict level):
  // fc_b, fc_w, layers[0..2]{ mlp1{b1,b2,w1,w2}, mlp2{b1,b2,w1,w2} }
  struct Mlp { const float *b1, *b2, *w1, *w2; };
  struct Gin { const float *fcb, *fcw; Mlp m1[3], m2[3]; };
  Gin g[2];
  int ix = 4;
  for (int gi = 0; gi < 2; ++gi) {
    g[gi].fcb = (const float*)d_in[ix++];
    g[gi].fcw = (const float*)d_in[ix++];
    for (int L = 0; L < 3; ++L) {
      g[gi].m1[L].b1 = (const float*)d_in[ix++];
      g[gi].m1[L].b2 = (const float*)d_in[ix++];
      g[gi].m1[L].w1 = (const float*)d_in[ix++];
      g[gi].m1[L].w2 = (const float*)d_in[ix++];
      g[gi].m2[L].b1 = (const float*)d_in[ix++];
      g[gi].m2[L].b2 = (const float*)d_in[ix++];
      g[gi].m2[L].w1 = (const float*)d_in[ix++];
      g[gi].m2[L].w2 = (const float*)d_in[ix++];
    }
  }

  char* ws = (char*)d_ws;
  int*      knn = (int*)(ws + OFF_KNN);
  _Float16* xh0 = (_Float16*)(ws + OFF_XH0);
  _Float16* xA  = (_Float16*)(ws + OFF_XA);
  _Float16* xB  = (_Float16*)(ws + OFF_XB);
  float*    agg = (float*)(ws + OFF_AGG);

  // per-(gin,layer) swizzled weight block: w1m1s, w2m1s, w1m2s, w2m2s, cvec
  auto wbase = [&](int gi, int L) { return ws + OFF_W + (size_t)(gi * 3 + L) * 32768; };

  // ---- stage 0: weight prep + features + knn --------------------------------
  for (int gi = 0; gi < 2; ++gi)
    for (int L = 0; L < 3; ++L) {
      const int din = (L == 0) ? 35 : 32;
      char* wb = wbase(gi, L);
      prep_layer_kernel<<<1, 256, 0, stream>>>(
          g[gi].m1[L].w1, g[gi].m1[L].b1, g[gi].m1[L].w2, g[gi].m1[L].b2,
          g[gi].m2[L].w1, g[gi].m2[L].w2, din,
          (_Float16*)(wb + 0),     (_Float16*)(wb + 8192),
          (_Float16*)(wb + 16384), (_Float16*)(wb + 24576),
          (float*)(wb + 28672));
    }

  build_x_kernel<<<(BB * NN * XSTR + 255) / 256, 256, 0, stream>>>(pos, atoms, emb, xh0);
  knn_kernel<<<dim3(NN / 256, BB), 256, 0, stream>>>(pos, knn);

  // ---- stage 1: two GIN stacks ---------------------------------------------
  for (int gi = 0; gi < 2; ++gi) {
    const _Float16* xin = xh0;
    _Float16* xnext = xA;
    for (int L = 0; L < 3; ++L) {
      const int din = (L == 0) ? 35 : 32;
      char* wb = wbase(gi, L);
      const _Float16* w1m1s = (const _Float16*)(wb + 0);
      const _Float16* w2m1s = (const _Float16*)(wb + 8192);
      const _Float16* w1m2s = (const _Float16*)(wb + 16384);
      const _Float16* w2m2s = (const _Float16*)(wb + 24576);
      const float*    cvec  = (const float*)(wb + 28672);

      init_agg_kernel<<<(BB * NN * XSTR + 255) / 256, 256, 0, stream>>>(agg, cvec, din);

      const int etiles = (gi == 0) ? (NN / 16) : (NN * KNNK / 16);
      const dim3 egrid((etiles + 7) / 8, BB);
      if (gi == 0) {
        if (din == 35)
          gin_edge_kernel<35, false><<<egrid, 256, 0, stream>>>(
              xin, knn, bonds, etiles, w1m1s, g[gi].m1[L].b1, w2m1s, g[gi].m1[L].b2, agg);
        else
          gin_edge_kernel<32, false><<<egrid, 256, 0, stream>>>(
              xin, knn, bonds, etiles, w1m1s, g[gi].m1[L].b1, w2m1s, g[gi].m1[L].b2, agg);
      } else {
        if (din == 35)
          gin_edge_kernel<35, true><<<egrid, 256, 0, stream>>>(
              xin, knn, bonds, etiles, w1m1s, g[gi].m1[L].b1, w2m1s, g[gi].m1[L].b2, agg);
        else
          gin_edge_kernel<32, true><<<egrid, 256, 0, stream>>>(
              xin, knn, bonds, etiles, w1m1s, g[gi].m1[L].b1, w2m1s, g[gi].m1[L].b2, agg);
      }

      const dim3 ngrid(NN / 16 / 8, BB);
      if (din == 35)
        gin_node_kernel<35><<<ngrid, 256, 0, stream>>>(
            agg, w1m2s, g[gi].m2[L].b1, w2m2s, g[gi].m2[L].b2, xnext);
      else
        gin_node_kernel<32><<<ngrid, 256, 0, stream>>>(
            agg, w1m2s, g[gi].m2[L].b1, w2m2s, g[gi].m2[L].b2, xnext);

      xin = xnext;
      xnext = (xnext == xA) ? xB : xA;
    }
    if (gi == 0)
      fc_kernel<false><<<(BB * NN + 255) / 256, 256, 0, stream>>>(
          xin, g[gi].fcw, g[gi].fcb, (float*)d_out);
    else
      fc_kernel<true><<<(BB * NN + 255) / 256, 256, 0, stream>>>(
          xin, g[gi].fcw, g[gi].fcb, (float*)d_out);
  }
}
